// MultiheadSelfAttention_16346645529343
// MI455X (gfx1250) — matrix operand order repaired and checked by reading, not compiled, not run on hardware
//
#include <hip/hip_runtime.h>

typedef __attribute__((ext_vector_type(16))) __bf16 v16bf;
typedef __attribute__((ext_vector_type(8)))  float  v8f;

#define WMMA_BF16(a, b, c) \
  __builtin_amdgcn_wmma_f32_16x16x32_bf16(false, (a), false, (b), (short)0, (c), false, false)

// ---- problem constants ----
#define BATCH 2
#define SEQ   2048
#define DMODEL 2048
#define NHEAD 16
#define DHEAD 128
#define NTOK  (BATCH * SEQ)     // 4096
#define QKVN  (3 * DMODEL)      // 6144

// Load a v16bf fragment from two 16-byte chunks (each 8 bf16, 16B aligned).
static __device__ __forceinline__ v16bf load2(const __bf16* a, const __bf16* b) {
  union { v16bf v; uint4 q[2]; } u;
  u.q[0] = *(const uint4*)a;
  u.q[1] = *(const uint4*)b;
  return u.v;
}

// ---------------- cast f32 -> bf16 ----------------
__global__ void k_cast_bf16(const float* __restrict__ s, __bf16* __restrict__ d, int n) {
  int i = blockIdx.x * blockDim.x + threadIdx.x;
  if (i < n) d[i] = (__bf16)s[i];
}

// ---------------- WMMA GEMM: C[M,N] = A[M,K] * W[N,K]^T ----------------
// Tiles: 128(M) x 128(N) x 32(K), ping-pong LDS, one barrier per K-step.
// 256 threads = 8 waves in a 4(M) x 2(N) grid; each wave: 32x64 = 2x4 C frags.
template <bool OUT_F32>
__global__ __launch_bounds__(256) void k_gemm_bf16(const __bf16* __restrict__ A,
                                                   const __bf16* __restrict__ W,
                                                   void* __restrict__ Cout,
                                                   int M, int N, int K) {
  __shared__ __bf16 As[2][128 * 40];   // 32 data + 8 pad per row
  __shared__ __bf16 Bs[2][128 * 40];

  const int tid  = threadIdx.x;
  const int wid  = tid >> 5;
  const int lane = tid & 31;
  const int lo   = lane & 15;
  const int hi   = lane >> 4;          // 0 or 1
  const int wm   = wid >> 1;           // 0..3  -> 32-row band
  const int wn   = wid & 1;            // 0..1  -> 64-col band
  const int gm0  = blockIdx.y * 128;
  const int gn0  = blockIdx.x * 128;

  v8f acc[2][4];
#pragma unroll
  for (int i = 0; i < 2; ++i)
#pragma unroll
    for (int j = 0; j < 4; ++j)
#pragma unroll
      for (int r = 0; r < 8; ++r) acc[i][j][r] = 0.0f;

  const int srow  = tid >> 1;          // 0..127
  const int shalf = (tid & 1) * 16;    // element offset 0 or 16
  const __bf16* gA = A + (size_t)(gm0 + srow) * K + shalf;
  const __bf16* gW = W + (size_t)(gn0 + srow) * K + shalf;

  // prologue: fetch tile 0 into registers
  uint4 ra0 = ((const uint4*)gA)[0], ra1 = ((const uint4*)gA)[1];
  uint4 rb0 = ((const uint4*)gW)[0], rb1 = ((const uint4*)gW)[1];

  const int niter = K / 32;
  for (int i = 0; i < niter; ++i) {
    __bf16* as = As[i & 1];
    __bf16* bs = Bs[i & 1];
    // commit staged tile to this iteration's LDS buffer
    uint4* sa = (uint4*)(as + srow * 40 + shalf);
    uint4* sb = (uint4*)(bs + srow * 40 + shalf);
    sa[0] = ra0; sa[1] = ra1;
    sb[0] = rb0; sb[1] = rb1;
    __syncthreads();

    // prefetch next tile (overlaps with the WMMA work below)
    if (i + 1 < niter) {
      const uint4* ga = (const uint4*)(gA + (i + 1) * 32);
      const uint4* gb = (const uint4*)(gW + (i + 1) * 32);
      ra0 = ga[0]; ra1 = ga[1];
      rb0 = gb[0]; rb1 = gb[1];
    }

    // fragments + matrix math
    v16bf af[2];
#pragma unroll
    for (int mf = 0; mf < 2; ++mf) {
      const __bf16* base = as + (wm * 32 + mf * 16 + lo) * 40;
      af[mf] = load2(base + hi * 16, base + hi * 16 + 8);   // A layout: K 0-15 / 16-31
    }
    v16bf bfr[4];
#pragma unroll
    for (int nf = 0; nf < 4; ++nf) {
      const __bf16* base = bs + (wn * 64 + nf * 16 + lo) * 40 + hi * 16;
      bfr[nf] = load2(base, base + 8);                     // B layout: K 0-15 / 16-31
    }
#pragma unroll
    for (int mf = 0; mf < 2; ++mf)
#pragma unroll
      for (int nf = 0; nf < 4; ++nf)
        acc[mf][nf] = WMMA_BF16(af[mf], bfr[nf], acc[mf][nf]);
  }

  // ---- epilogue: C layout -> row (VGPR r + 8*hi), col = lane lo ----
#pragma unroll
  for (int mf = 0; mf < 2; ++mf) {
    const int row0 = gm0 + wm * 32 + mf * 16 + hi * 8;
#pragma unroll
    for (int nf = 0; nf < 4; ++nf) {
      const int col = gn0 + wn * 64 + nf * 16 + lo;
#pragma unroll
      for (int r = 0; r < 8; ++r) {
        const size_t idx = (size_t)(row0 + r) * N + col;
        if (OUT_F32) ((float*)Cout)[idx] = acc[mf][nf][r];
        else         ((__bf16*)Cout)[idx] = (__bf16)acc[mf][nf][r];
      }
    }
  }
}

// ---------------- RoPE in place on Q and K regions of qkv ----------------
__global__ void k_rope(__bf16* __restrict__ qkv, const int* __restrict__ tp) {
  const int idx = blockIdx.x * blockDim.x + threadIdx.x;   // NTOK * 2048 pairs
  const int t = idx >> 11;
  const int c = idx & 2047;
  const int region = c >> 10;        // 0 = Q, 1 = K
  const int p = c & 1023;            // pair index within region
  const int i = p & 63;              // pair index within head (dh/2 = 64)
  const int col = region * DMODEL + 2 * p;
  const float pos = (float)tp[t & (SEQ - 1)];
  const float freq = __powf(10000.0f, -(float)i * (1.0f / 64.0f));
  float s, cz;
  __sincosf(pos * freq, &s, &cz);
  __bf16* ptr = qkv + (size_t)t * QKVN + col;
  const float x1 = (float)ptr[0];
  const float x2 = (float)ptr[1];
  ptr[0] = (__bf16)(cz * x1 - s * x2);
  ptr[1] = (__bf16)(s * x1 + cz * x2);
}

// ---------------- flash attention (causal) ----------------
// grid: (S/128, B*H). 256 threads = 8 waves, wave w owns queries [Q0+16w, Q0+16w+16).
// Scores computed transposed (S^T = K * Q^T): softmax stats are per-lane and the
// exp'd C-fragments are directly in WMMA A-operand layout for the P*V product.
// K (row-major) and V (transposed) are cooperatively staged into ping-pong LDS;
// the next 32-key block is prefetched into registers during the current math.
__global__ __launch_bounds__(256) void k_attn(const __bf16* __restrict__ qkv,
                                              __bf16* __restrict__ ao) {
  __shared__ __bf16 Kt[2][32 * 136];     // K block: [key][128 feats + 8 pad]
  __shared__ __bf16 Vt[2][DHEAD * 40];   // V transposed: [feat][32 keys + 8 pad]

  const int Q0   = blockIdx.x * 128;
  const int bh   = blockIdx.y;
  const int b    = bh >> 4;
  const int h    = bh & 15;
  const int tid  = threadIdx.x;
  const int wid  = tid >> 5;
  const int lane = tid & 31;
  const int lo   = lane & 15;
  const int hi   = lane >> 4;
  const int qw   = Q0 + wid * 16;                // wave query base
  const size_t tokb = (size_t)b * SEQ;

  // Q fragments as the B operand of S^T = K * Q^T (lane = query column).
  v16bf qf[4];
#pragma unroll
  for (int f = 0; f < 4; ++f) {
    const __bf16* qp = qkv + (tokb + qw + lo) * QKVN + h * DHEAD + f * 32 + hi * 8;
    qf[f] = load2(qp, qp + 16);
  }

  v8f o[8];
#pragma unroll
  for (int nf = 0; nf < 8; ++nf)
#pragma unroll
    for (int r = 0; r < 8; ++r) o[nf][r] = 0.0f;

  float mrow = -1e30f, lrow = 0.0f;              // stats for query qw+lo (lanes lo & lo+16)
  const float scale = 0.0883883476483184f;       // 1/sqrt(128)
  const int nkb = (Q0 + 128) / 32;

  // K staging: thread handles key row tid>>3, feats [(tid&7)*16, +16)
  const int kkey = tid >> 3;
  const int kcol = (tid & 7) * 16;
  const __bf16* ksrc = qkv + (tokb + kkey) * QKVN + DMODEL + h * DHEAD + kcol;
  uint4 uk0 = ((const uint4*)ksrc)[0];           // prologue: K block 0
  uint4 uk1 = ((const uint4*)ksrc)[1];

  // V staging: thread handles key (tid&31), feats [(tid>>5)*16, +16), transposed
  const int vkey = tid & 31;
  const int vf0  = (tid >> 5) * 16;
  const __bf16* vsrc = qkv + (tokb + vkey) * QKVN + 2 * DMODEL + h * DHEAD + vf0;
  union { uint4 q[2]; __bf16 e[16]; } u;
  u.q[0] = *(const uint4*)vsrc;                  // prologue: V block 0
  u.q[1] = *(const uint4*)(vsrc + 8);

  for (int kb = 0; kb < nkb; ++kb) {
    const int kbase = kb * 32;
    __bf16* kt = Kt[kb & 1];
    __bf16* vt = Vt[kb & 1];

    // commit staged K (row-major) and V (transposed) blocks
    {
      uint4* sk = (uint4*)(kt + kkey * 136 + kcol);
      sk[0] = uk0; sk[1] = uk1;
    }
#pragma unroll
    for (int c = 0; c < 16; ++c) vt[(vf0 + c) * 40 + vkey] = u.e[c];
    __syncthreads();

    // prefetch next blocks (overlaps with the math below)
    if (kb + 1 < nkb) {
      const __bf16* kp = ksrc + (size_t)(kbase + 32) * QKVN;
      uk0 = ((const uint4*)kp)[0];
      uk1 = ((const uint4*)kp)[1];
      const __bf16* vp = vsrc + (size_t)(kbase + 32) * QKVN;
      u.q[0] = *(const uint4*)vp;
      u.q[1] = *(const uint4*)(vp + 8);
    }

    if (kbase <= qw + 15) {  // wave-uniform: some key in this block is unmasked
      // ---- S^T = K_block * Q^T  (two 16-key fragments, two accum chains) ----
      v8f s0, s1;
#pragma unroll
      for (int r = 0; r < 8; ++r) { s0[r] = 0.0f; s1[r] = 0.0f; }
#pragma unroll
      for (int f = 0; f < 4; ++f) {
        const __bf16* r0 = kt + lo * 136 + f * 32;          // keys kbase+0..15
        const __bf16* r1 = kt + (16 + lo) * 136 + f * 32;   // keys kbase+16..31
        v16bf ka0 = load2(r0 + hi * 8, r0 + 16 + hi * 8);
        v16bf ka1 = load2(r1 + hi * 8, r1 + 16 + hi * 8);
        s0 = WMMA_BF16(ka0, qf[f], s0);
        s1 = WMMA_BF16(ka1, qf[f], s1);
      }

      // ---- scale + causal mask + online softmax (per-lane query stats) ----
      const int q_g  = qw + lo;
      const int k0_g = kbase + 8 * hi;
      float e0[8], e1[8];
      float mb = -1e30f;
#pragma unroll
      for (int r = 0; r < 8; ++r) {
        float v0 = s0[r] * scale; if (k0_g + r > q_g)      v0 = -1e30f;
        float v1 = s1[r] * scale; if (k0_g + 16 + r > q_g) v1 = -1e30f;
        e0[r] = v0; e1[r] = v1;
        mb = fmaxf(mb, fmaxf(v0, v1));
      }
      mb = fmaxf(mb, __shfl_xor(mb, 16, 32));
      const float mnew = fmaxf(mrow, mb);
      const float corr = __expf(mrow - mnew);
      float rsum = 0.0f;
#pragma unroll
      for (int r = 0; r < 8; ++r) {
        e0[r] = __expf(e0[r] - mnew);
        e1[r] = __expf(e1[r] - mnew);
        rsum += e0[r] + e1[r];
      }
      rsum += __shfl_xor(rsum, 16, 32);
      lrow = lrow * corr + rsum;
      mrow = mnew;

      // rescale accumulator: query of O-row r is qw + r + 8*hi -> stats on lane r+8*hi
      float cb[8];
#pragma unroll
      for (int r = 0; r < 8; ++r) cb[r] = __shfl(corr, r + 8 * hi, 32);
#pragma unroll
      for (int nf = 0; nf < 8; ++nf)
#pragma unroll
        for (int r = 0; r < 8; ++r) o[nf][r] *= cb[r];

      // ---- P A-fragment: exp'd S^T C-layout is already A-operand layout ----
      union { v16bf v; __bf16 e[16]; } pu;
#pragma unroll
      for (int r = 0; r < 8; ++r) {
        pu.e[r]     = (__bf16)e0[r];
        pu.e[8 + r] = (__bf16)e1[r];
      }

      // ---- O += P * V  (V B-fragments from transposed LDS tile) ----
#pragma unroll
      for (int nf = 0; nf < 8; ++nf) {
        const __bf16* vb = vt + (nf * 16 + lo) * 40 + hi * 8;
        v16bf vf = load2(vb, vb + 16);
        o[nf] = WMMA_BF16(pu.v, vf, o[nf]);
      }
    }
  }

  // ---- normalize and store (row-major [token][D] for the output GEMM) ----
  const float li = 1.0f / lrow;
  float lb[8];
#pragma unroll
  for (int r = 0; r < 8; ++r) lb[r] = __shfl(li, r + 8 * hi, 32);
#pragma unroll
  for (int nf = 0; nf < 8; ++nf) {
    const int feat = nf * 16 + lo;
#pragma unroll
    for (int r = 0; r < 8; ++r) {
      const int q = qw + r + 8 * hi;
      ao[(tokb + q) * DMODEL + h * DHEAD + feat] = (__bf16)(o[nf][r] * lb[r]);
    }
  }
}

// ---------------- launcher ----------------
extern "C" void kernel_launch(void* const* d_in, const int* in_sizes, int n_in,
                              void* d_out, int out_size, void* d_ws, size_t ws_size,
                              hipStream_t stream) {
  const float* x  = (const float*)d_in[0];
  const float* Wq = (const float*)d_in[1];
  const float* Wk = (const float*)d_in[2];
  const float* Wv = (const float*)d_in[3];
  const float* Wo = (const float*)d_in[4];
  const int*   tp = (const int*)d_in[5];

  // workspace layout (bf16 elements)
  __bf16* xb   = (__bf16*)d_ws;                         //  4096 x 2048
  __bf16* wqkv = xb + (size_t)NTOK * DMODEL;            //  6144 x 2048  (Wq;Wk;Wv)
  __bf16* wo   = wqkv + (size_t)QKVN * DMODEL;          //  2048 x 2048
  __bf16* qkvb = wo + (size_t)DMODEL * DMODEL;          //  4096 x 6144
  __bf16* ao   = qkvb + (size_t)NTOK * QKVN;            //  4096 x 2048

  const int nx = NTOK * DMODEL;        // 8,388,608
  const int nw = DMODEL * DMODEL;      // 4,194,304

  // 1) downcast inputs
  k_cast_bf16<<<(nx + 255) / 256, 256, 0, stream>>>(x, xb, nx);
  k_cast_bf16<<<(nw + 255) / 256, 256, 0, stream>>>(Wq, wqkv,          nw);
  k_cast_bf16<<<(nw + 255) / 256, 256, 0, stream>>>(Wk, wqkv + nw,     nw);
  k_cast_bf16<<<(nw + 255) / 256, 256, 0, stream>>>(Wv, wqkv + 2 * nw, nw);
  k_cast_bf16<<<(nw + 255) / 256, 256, 0, stream>>>(Wo, wo,            nw);

  // 2) fused QKV projection: qkv = x @ [Wq;Wk;Wv]^T   (4096 x 6144)
  k_gemm_bf16<false><<<dim3(QKVN / 128, NTOK / 128), 256, 0, stream>>>(
      xb, wqkv, (void*)qkvb, NTOK, QKVN, DMODEL);

  // 3) RoPE on Q and K regions
  k_rope<<<(NTOK * 2048) / 256, 256, 0, stream>>>(qkvb, tp);

  // 4) causal flash attention per (b,h) x 128-query tile
  k_attn<<<dim3(SEQ / 128, BATCH * NHEAD), 256, 0, stream>>>(qkvb, ao);

  // 5) output projection: out = ao @ Wo^T  (f32 result)
  k_gemm_bf16<true><<<dim3(DMODEL / 128, NTOK / 128), 256, 0, stream>>>(
      ao, wo, d_out, NTOK, DMODEL, DMODEL);
}
